// MultiheadAttention_23270132810264
// MI455X (gfx1250) — matrix operand order repaired and checked by reading, not compiled, not run on hardware
//
#include <hip/hip_runtime.h>
#include <hip/hip_bf16.h>

// ---------------------------------------------------------------------------
// Multi-head attention forward, MI455X (gfx1250, wave32, WMMA + TDM).
//   B=4, S=2048, D=1024, H=16, dk=64
// Pipeline: 3x proj GEMM (f32 in -> bf16 head-split out, ReLU) ->
//           flash attention (bf16 WMMA, f32 online softmax, TDM K-tiles) ->
//           output GEMM (bf16 in via TDM -> f32 out, ReLU).
// ---------------------------------------------------------------------------

#define Dm 1024
#define Hh 16
#define Bb 4
#define Ss 2048
#define DKq 64

typedef __attribute__((ext_vector_type(16))) __bf16 v16bf;
typedef __attribute__((ext_vector_type(8)))  float  v8f;
typedef __attribute__((ext_vector_type(4)))  unsigned int u32x4;
typedef __attribute__((ext_vector_type(8)))  int i32x8;
typedef __attribute__((ext_vector_type(4)))  int i32x4;

#if __has_builtin(__builtin_amdgcn_tensor_load_to_lds)
#define USE_TDM 1
#else
#define USE_TDM 0
#endif

union Frag {
  uint4 q[2];
  v16bf v;
};

__device__ __forceinline__ unsigned short f2bf(float f) {
  unsigned u = __builtin_bit_cast(unsigned, f);
  u += 0x7FFFu + ((u >> 16) & 1u);            // round to nearest even
  return (unsigned short)(u >> 16);
}

__device__ __forceinline__ void wait_tensorcnt0() {
#if __has_builtin(__builtin_amdgcn_s_wait_tensorcnt)
  __builtin_amdgcn_s_wait_tensorcnt(0);
#else
  asm volatile("s_wait_tensorcnt 0x0" ::: "memory");
#endif
}

#if USE_TDM
// 2D bf16 tile DMA: global (row pitch = row_stride_elems) -> LDS with row
// padding (pad codes per ISA D# group1: interval 0=2dw..4=32dw, amount 0=1dw..).
// Tensor dims are set equal to tile dims (tiles never out-of-bounds here);
// the real tensor pitch goes in tensor_dim0_stride.
__device__ __forceinline__ void tdm_load_2d_bf16(
    unsigned lds_off, const void* gtile, unsigned tile_w, unsigned tile_h,
    unsigned row_stride_elems, unsigned pad_interval_code,
    unsigned pad_amount_code) {
  unsigned long long ga = (unsigned long long)(uintptr_t)gtile;
  u32x4 g0 = {
      1u,                                            // count=1, user mode
      lds_off,                                       // lds_addr
      (unsigned)ga,                                  // global_addr[31:0]
      (unsigned)((ga >> 32) & 0x01FFFFFFull) | (2u << 30)  // addr[56:32]|type=2
  };
  i32x8 g1 = {
      (int)((1u << 16) |                             // data_size = 2 bytes
            (1u << 20) |                             // pad_enable
            (pad_interval_code << 22) | (pad_amount_code << 25)),
      (int)(tile_w << 16),                           // tensor_dim0[15:0]
      (int)((tile_w >> 16) | (tile_h << 16)),        // dim0 hi | tensor_dim1 lo
      (int)((tile_h >> 16) | (tile_w << 16)),        // dim1 hi | tile_dim0
      (int)tile_h,                                   // tile_dim1 (tile_dim2=0)
      (int)row_stride_elems,                         // tensor_dim0_stride lo
      0, 0};
  i32x4 z4 = {0, 0, 0, 0};
#if defined(__clang_major__) && (__clang_major__ >= 23)
  i32x8 z8 = {0, 0, 0, 0, 0, 0, 0, 0};
  __builtin_amdgcn_tensor_load_to_lds(g0, g1, z4, z4, z8, 0);
#else
  __builtin_amdgcn_tensor_load_to_lds(g0, g1, z4, z4, 0);
#endif
}
#endif

// ---------------------------------------------------------------------------
// GEMM: C[M=8192, N=1024] = ReLU(X @ W + bias)
//   IN_BF16:     X is bf16 (A tile staged by TDM); else f32 converted in VALU
//   SPLIT_HEADS: write bf16 into [B][H][S][64] layout (else f32 row-major)
// Block: 256 threads (8 waves), tile 128(M) x 64(N), K step 32.
// ---------------------------------------------------------------------------
template <bool IN_BF16, bool SPLIT_HEADS>
__global__ __launch_bounds__(256) void proj_gemm(
    const void* __restrict__ Xv, const float* __restrict__ W,
    const float* __restrict__ bias, void* __restrict__ outv) {
  __shared__ unsigned short lA[128][40] __attribute__((aligned(16)));  // [m][k]
  __shared__ unsigned short lB[64][40]  __attribute__((aligned(16)));  // [n][k]

  const int t    = threadIdx.x;
  const int wave = t >> 5;
  const int lane = t & 31;
  const int hi   = lane >> 4;   // which 16-lane half
  const int l16  = lane & 15;
  const int m0   = blockIdx.y * 128;
  const int n0   = blockIdx.x * 64;

  v8f acc[4] = {};

  for (int k0 = 0; k0 < Dm; k0 += 32) {
    // ---- stage A tile 128x32 ----
    if (!IN_BF16) {
      const float* X = (const float*)Xv;
#pragma unroll
      for (int i = 0; i < 4; ++i) {
        int e = t + i * 256;                 // 1024 float4 chunks
        int row = e >> 3, c4 = e & 7;
        const float4 f =
            *(const float4*)(X + (size_t)(m0 + row) * Dm + k0 + c4 * 4);
        unsigned lo = (unsigned)f2bf(f.x) | ((unsigned)f2bf(f.y) << 16);
        unsigned h2 = (unsigned)f2bf(f.z) | ((unsigned)f2bf(f.w) << 16);
        *(uint2*)&lA[row][c4 * 4] = make_uint2(lo, h2);
      }
    } else {
#if USE_TDM
      // Tensor Data Mover: 128x32 bf16 tile, LDS row stride 40 elems
      // (pad_interval 16 dwords -> code 3, pad_amount 4 dwords -> code 3).
      if (wave == 0)
        tdm_load_2d_bf16((unsigned)(uintptr_t)&lA[0][0],
                         (const unsigned short*)Xv + (size_t)m0 * Dm + k0,
                         32, 128, Dm, 3, 3);
#else
      const unsigned short* X = (const unsigned short*)Xv;
#pragma unroll
      for (int i = 0; i < 4; ++i) {
        int e = t + i * 256;
        int row = e >> 3, c4 = e & 7;
        uint2 u = *(const uint2*)(X + (size_t)(m0 + row) * Dm + k0 + c4 * 4);
        *(uint2*)&lA[row][c4 * 4] = u;
      }
#endif
    }
    // ---- stage B tile 32x64, transposed into [n][k], f32->bf16 ----
#pragma unroll
    for (int i = 0; i < 8; ++i) {
      int e = t + i * 256;                   // 2048 elements
      int kk = e >> 6, n = e & 63;
      lB[n][kk] = f2bf(W[(size_t)(k0 + kk) * Dm + n0 + n]);
    }
    // prefetch next K-slice of W (global_prefetch_b8)
    if (k0 + 32 < Dm)
      __builtin_prefetch(&W[(size_t)(k0 + 32 + (t >> 3)) * Dm + n0 + (t & 7) * 8], 0, 0);
#if USE_TDM
    if (IN_BF16 && wave == 0) wait_tensorcnt0();
#endif
    __syncthreads();

    // ---- WMMA: wave owns rows [wave*16, wave*16+16), all 64 N columns ----
    Frag a;
    const int arow = wave * 16 + l16;
    a.q[0] = *(const uint4*)&lA[arow][hi * 8];        // K = base..base+7
    a.q[1] = *(const uint4*)&lA[arow][hi * 8 + 16];   // K = 16+base..
#pragma unroll
    for (int j = 0; j < 4; ++j) {
      Frag b;
      const int bn = j * 16 + l16;
      b.q[0] = *(const uint4*)&lB[bn][hi * 8];        // K = base..base+7
      b.q[1] = *(const uint4*)&lB[bn][hi * 8 + 16];
      acc[j] = __builtin_amdgcn_wmma_f32_16x16x32_bf16(
          false, a.v, false, b.v, (short)0, acc[j], false, false);
    }
    __syncthreads();
  }

  // ---- epilogue: bias + ReLU, scatter per C-layout ----
#pragma unroll
  for (int j = 0; j < 4; ++j) {
    const int n = n0 + j * 16 + l16;
    const float bn = bias[n];
#pragma unroll
    for (int r = 0; r < 8; ++r) {
      const int m = m0 + wave * 16 + r + hi * 8;
      float v = acc[j][r] + bn;
      v = v > 0.f ? v : 0.f;
      if (SPLIT_HEADS) {
        unsigned short* out = (unsigned short*)outv;
        int bb = m >> 11, s = m & 2047, h = n >> 6, d = n & 63;
        out[((((size_t)bb * Hh + h) * Ss) + s) * DKq + d] = f2bf(v);
      } else {
        float* out = (float*)outv;
        out[(size_t)m * Dm + n] = v;
      }
    }
  }
}

// ---------------------------------------------------------------------------
// Flash attention over one (b,h) head slice: Q,K,V are bf16 [S][64].
// Block: 256 threads (8 waves) handles a 128-query tile; each wave owns 16
// query rows, keeps Q fragments in registers, iterates 64-key tiles.
// K tile staged by the Tensor Data Mover (overlaps with V transpose staging).
// Output written bf16 into Y[B][S][D] (col = h*64+d) for the final GEMM.
// ---------------------------------------------------------------------------
__global__ __launch_bounds__(256) void flash_attn(
    const unsigned short* __restrict__ Qh, const unsigned short* __restrict__ Kh,
    const unsigned short* __restrict__ Vh, unsigned short* __restrict__ Y) {
  __shared__ unsigned short kT[64][72] __attribute__((aligned(16)));      // [kv][d]
  __shared__ unsigned short vT[64][72] __attribute__((aligned(16)));      // [d][kv]
  __shared__ unsigned short pT[8][16][72] __attribute__((aligned(16)));   // per-wave P

  const int t    = threadIdx.x;
  const int wave = t >> 5;
  const int lane = t & 31;
  const int hi   = lane >> 4;
  const int l16  = lane & 15;
  const int bh   = blockIdx.y;          // b*H + h
  const int q0   = blockIdx.x * 128;
  const size_t base = (size_t)bh * Ss * DKq;
  const unsigned short* Q = Qh + base;
  const unsigned short* K = Kh + base;
  const unsigned short* V = Vh + base;

  // Q fragments: 16 rows x dk=64 -> two 16x32 bf16 A-fragments, in registers
  Frag qf[2];
  {
    const int row = q0 + wave * 16 + l16;
#pragma unroll
    for (int c = 0; c < 2; ++c) {
      const unsigned short* p = Q + (size_t)row * DKq + c * 32 + hi * 8;
      qf[c].q[0] = *(const uint4*)p;
      qf[c].q[1] = *(const uint4*)(p + 16);
    }
  }

  float rowmax[8], rowsum[8];
#pragma unroll
  for (int r = 0; r < 8; ++r) { rowmax[r] = -1e30f; rowsum[r] = 0.f; }
  v8f yacc[4] = {};
  const float scale = 0.125f;           // 1/sqrt(dk)

  for (int kv0 = 0; kv0 < Ss; kv0 += 64) {
    // ---- stage K tile 64x64 row-major via TDM (LDS row stride 72 elems:
    //      pad_interval 32 dwords -> code 4, pad_amount 4 dwords -> code 3) ----
#if USE_TDM
    if (wave == 0)
      tdm_load_2d_bf16((unsigned)(uintptr_t)&kT[0][0],
                       K + (size_t)kv0 * DKq, 64, 64, DKq, 4, 3);
#else
#pragma unroll
    for (int i = 0; i < 2; ++i) {
      int c8 = t + i * 256;             // 512 chunks of 8 ushorts
      int row = c8 >> 3, off = (c8 & 7) * 8;
      *(uint4*)&kT[row][off] =
          *(const uint4*)(K + (size_t)(kv0 + row) * DKq + off);
    }
#endif
    // ---- stage V tile transposed [d][kv] (VALU path, overlaps TDM) ----
#pragma unroll
    for (int i = 0; i < 8; ++i) {
      int e2 = t + i * 256;             // 2048 uint chunks (2 bf16 each)
      int kv = e2 >> 5, d2 = (e2 & 31) * 2;
      unsigned u = *(const unsigned*)(V + (size_t)(kv0 + kv) * DKq + d2);
      vT[d2][kv]     = (unsigned short)(u & 0xFFFFu);
      vT[d2 + 1][kv] = (unsigned short)(u >> 16);
    }
    if (kv0 + 64 < Ss)
      __builtin_prefetch(V + (size_t)(kv0 + 64 + (t >> 2)) * DKq + (t & 3) * 16, 0, 0);
#if USE_TDM
    if (wave == 0) wait_tensorcnt0();
#endif
    __syncthreads();

    // ---- scores S = Q @ K^T (16 q-rows x 64 kv-cols per wave) ----
    v8f sc[4] = {};
#pragma unroll
    for (int j = 0; j < 4; ++j) {
#pragma unroll
      for (int c = 0; c < 2; ++c) {
        Frag b;
        const int kv = j * 16 + l16;
        const int db = c * 32 + hi * 8;
        b.q[0] = *(const uint4*)&kT[kv][db];
        b.q[1] = *(const uint4*)&kT[kv][db + 16];
        sc[j] = __builtin_amdgcn_wmma_f32_16x16x32_bf16(
            false, qf[c].v, false, b.v, (short)0, sc[j], false, false);
      }
    }

    // ---- online softmax (row lives on 16 lanes of one half-wave) ----
#pragma unroll
    for (int r = 0; r < 8; ++r) {
      float mx = fmaxf(fmaxf(sc[0][r], sc[1][r]), fmaxf(sc[2][r], sc[3][r]));
#pragma unroll
      for (int off = 1; off <= 8; off <<= 1)
        mx = fmaxf(mx, __shfl_xor(mx, off, 32));
      mx *= scale;
      float mn   = fmaxf(rowmax[r], mx);
      float corr = __expf(rowmax[r] - mn);
      rowmax[r]  = mn;
      float ps = 0.f;
#pragma unroll
      for (int j = 0; j < 4; ++j) {
        float p = __expf(sc[j][r] * scale - mn);
        sc[j][r] = p;
        ps += p;
      }
#pragma unroll
      for (int off = 1; off <= 8; off <<= 1)
        ps += __shfl_xor(ps, off, 32);
      rowsum[r] = rowsum[r] * corr + ps;
#pragma unroll
      for (int j = 0; j < 4; ++j) yacc[j][r] *= corr;
    }

    // ---- C-layout -> A-layout via wave-private LDS (bf16) ----
#pragma unroll
    for (int j = 0; j < 4; ++j)
#pragma unroll
      for (int r = 0; r < 8; ++r)
        pT[wave][r + hi * 8][j * 16 + l16] = f2bf(sc[j][r]);

    // ---- y += P @ V ----
#pragma unroll
    for (int c = 0; c < 2; ++c) {
      Frag a;
      const unsigned short* p = &pT[wave][l16][c * 32 + hi * 8];
      a.q[0] = *(const uint4*)p;
      a.q[1] = *(const uint4*)(p + 16);
#pragma unroll
      for (int j = 0; j < 4; ++j) {
        Frag b;
        const int d  = j * 16 + l16;
        const int kb = c * 32 + hi * 8;
        b.q[0] = *(const uint4*)&vT[d][kb];
        b.q[1] = *(const uint4*)&vT[d][kb + 16];
        yacc[j] = __builtin_amdgcn_wmma_f32_16x16x32_bf16(
            false, a.v, false, b.v, (short)0, yacc[j], false, false);
      }
    }
    __syncthreads();
  }

  // ---- epilogue: normalize, write bf16 Y[b][s][h*64+d] ----
  const int b = bh >> 4, h = bh & 15;
#pragma unroll
  for (int j = 0; j < 4; ++j) {
#pragma unroll
    for (int r = 0; r < 8; ++r) {
      const int s   = q0 + wave * 16 + r + hi * 8;
      const int col = h * 64 + j * 16 + l16;
      float v = yacc[j][r] / rowsum[r];
      Y[((size_t)b * Ss + s) * Dm + col] = f2bf(v);
    }
  }
}

// ---------------------------------------------------------------------------
extern "C" void kernel_launch(void* const* d_in, const int* in_sizes, int n_in,
                              void* d_out, int out_size, void* d_ws,
                              size_t ws_size, hipStream_t stream) {
  (void)in_sizes; (void)n_in; (void)out_size; (void)ws_size;
  const float* queries = (const float*)d_in[0];
  const float* keys    = (const float*)d_in[1];
  const float* values  = (const float*)d_in[2];
  const float* Wq = (const float*)d_in[3];  const float* bq = (const float*)d_in[4];
  const float* Wk = (const float*)d_in[5];  const float* bk = (const float*)d_in[6];
  const float* Wv = (const float*)d_in[7];  const float* bv = (const float*)d_in[8];
  const float* Wo = (const float*)d_in[9];  const float* bo = (const float*)d_in[10];

  // workspace: Qh | Kh | Vh (bf16 [B][H][S][64]) | Y (bf16 [B][S][D]) = 64 MB
  const size_t HSLICE = (size_t)Bb * Hh * Ss * DKq;   // 8M elems
  unsigned short* Qh = (unsigned short*)d_ws;
  unsigned short* Kh = Qh + HSLICE;
  unsigned short* Vh = Kh + HSLICE;
  unsigned short* Yb = Vh + HSLICE;

  dim3 blk(256);
  dim3 gproj(Dm / 64, (Bb * Ss) / 128);   // (16, 64)
  hipLaunchKernelGGL((proj_gemm<false, true>), gproj, blk, 0, stream,
                     (const void*)queries, Wq, bq, (void*)Qh);
  hipLaunchKernelGGL((proj_gemm<false, true>), gproj, blk, 0, stream,
                     (const void*)keys, Wk, bk, (void*)Kh);
  hipLaunchKernelGGL((proj_gemm<false, true>), gproj, blk, 0, stream,
                     (const void*)values, Wv, bv, (void*)Vh);

  dim3 gattn(Ss / 128, Bb * Hh);          // (16, 64)
  hipLaunchKernelGGL(flash_attn, gattn, blk, 0, stream, Qh, Kh, Vh, Yb);

  hipLaunchKernelGGL((proj_gemm<true, false>), gproj, blk, 0, stream,
                     (const void*)Yb, Wo, bo, d_out);
}
